// EchoStateNetwork_62715112456296
// MI455X (gfx1250) — matrix operand order repaired and checked by reading, not compiled, not run on hardware
//
#include <hip/hip_runtime.h>
#include <math.h>

typedef __attribute__((ext_vector_type(16))) _Float16 v16h;
typedef __attribute__((ext_vector_type(8)))  _Float16 v8h;
typedef __attribute__((ext_vector_type(8)))  float    v8f;

constexpr int kB = 64;     // batch
constexpr int kT = 256;    // timesteps
constexpr int kH = 2048;   // hidden
constexpr int kO = 128;    // output classes

// ---------------------------------------------------------------------------
// Pack a row-major f32 KxN weight matrix into f16 WMMA-B fragments.
// Fragment (kt, nt) covers K rows [kt*32, kt*32+32), N cols [nt*16, nt*16+16).
// Fragment storage: 512 halves, lane-major: frag[l][e] = W[kt*32 + 16*(e/8) + 8*(l/16) + e%8][nt*16 + l%16]
// so each lane's 32-byte chunk is contiguous -> coalesced 1KB fragment loads.
// ---------------------------------------------------------------------------
__global__ __launch_bounds__(256) void pack_frag_kernel(
    const float* __restrict__ W, _Float16* __restrict__ out, int N, int NT) {
  int i = blockIdx.x * blockDim.x + threadIdx.x;
  int e = i & 15;
  int l = (i >> 4) & 31;
  int fragIdx = i >> 9;
  int nt = fragIdx % NT;
  int kt = fragIdx / NT;
  int k = kt * 32 + ((e >> 3) << 4) + ((l >> 4) << 3) + (e & 7);
  int n = nt * 16 + (l & 15);
  out[i] = (_Float16)W[k * N + n];
}

__device__ __forceinline__ v16h make_a_frag(const _Float16* arow, int kt, int k0) {
  v8h lo = *(const v8h*)(arow + kt * 32 + k0);
  v8h hi = *(const v8h*)(arow + kt * 32 + k0 + 16);
  v16h a;
#pragma unroll
  for (int q = 0; q < 8; ++q) { a[q] = lo[q]; a[8 + q] = hi[q]; }
  return a;
}

// ---------------------------------------------------------------------------
// One ESN timestep: h_new = tanh(Wxh[x[:,t]] + bh + h_old @ Whh)
// Grid: 16 blocks x 256 threads = 128 waves. Wave = 16(M) x 64(N) strip.
// A operand = previous state slice (f16, row-major), B = pre-swizzled WhhFrag.
// ---------------------------------------------------------------------------
__global__ __launch_bounds__(256) void esn_step_kernel(
    const _Float16* __restrict__ zprev, _Float16* __restrict__ zout,
    const _Float16* __restrict__ WhhFrag,
    const int* __restrict__ x, const float* __restrict__ Wxh,
    const float* __restrict__ bh, float* __restrict__ hlast,
    int t, int first) {
  const int lane = threadIdx.x & 31;
  const int wid  = (blockIdx.x * blockDim.x + threadIdx.x) >> 5;  // 0..127
  const int mt   = wid & 3;    // M tile (batch rows / 16)
  const int ns   = wid >> 2;   // N strip 0..31 (64 cols each)
  const int hrow = mt * 16 + (lane & 15);
  const int k0   = (lane >= 16) ? 8 : 0;  // 16-bit A layout K phase per half-wave

  v8f c0, c1, c2, c3;
#pragma unroll
  for (int q = 0; q < 8; ++q) { c0[q] = 0.0f; c1[q] = 0.0f; c2[q] = 0.0f; c3[q] = 0.0f; }

  if (!first) {
    const _Float16* arow  = zprev + (size_t)hrow * kH;
    const _Float16* bbase = WhhFrag + ((size_t)ns * 4) * 512 + lane * 16;
#pragma unroll 1
    for (int kt = 0; kt < 64; ++kt) {
      // Prefetch next K-step's B fragments (unconditional; kt=63 touches the
      // adjacent WoutFrag region -> valid memory, speculative prefetch).
      __builtin_prefetch(bbase + (size_t)(kt + 1) * 128 * 512, 0, 1);

      v16h a  = make_a_frag(arow, kt, k0);
      const _Float16* bp = bbase + (size_t)kt * 128 * 512;
      v16h b0 = *(const v16h*)(bp + 0 * 512);
      v16h b1 = *(const v16h*)(bp + 1 * 512);
      v16h b2 = *(const v16h*)(bp + 2 * 512);
      v16h b3 = *(const v16h*)(bp + 3 * 512);
      c0 = __builtin_amdgcn_wmma_f32_16x16x32_f16(false, a, false, b0, (short)0, c0, false, false);
      c1 = __builtin_amdgcn_wmma_f32_16x16x32_f16(false, a, false, b1, (short)0, c1, false, false);
      c2 = __builtin_amdgcn_wmma_f32_16x16x32_f16(false, a, false, b2, (short)0, c2, false, false);
      c3 = __builtin_amdgcn_wmma_f32_16x16x32_f16(false, a, false, b3, (short)0, c3, false, false);
    }
  }

  // Epilogue: + one-hot gather drive + bias, tanh, store f16 state (and f32 h_last)
  float vacc[4][8];
#pragma unroll
  for (int q = 0; q < 8; ++q) {
    vacc[0][q] = c0[q]; vacc[1][q] = c1[q]; vacc[2][q] = c2[q]; vacc[3][q] = c3[q];
  }
#pragma unroll
  for (int r = 0; r < 8; ++r) {
    int m   = mt * 16 + r + ((lane >= 16) ? 8 : 0);  // batch row (D layout)
    int tok = x[m * kT + t];
#pragma unroll
    for (int j = 0; j < 4; ++j) {
      int n = ns * 64 + j * 16 + (lane & 15);
      float val = tanhf(vacc[j][r] + Wxh[(size_t)tok * kH + n] + bh[n]);
      zout[(size_t)m * kH + n] = (_Float16)val;
      if (hlast) hlast[(size_t)m * kH + n] = val;
    }
  }
}

// ---------------------------------------------------------------------------
// Readout: y = softmax(z @ W_out + b_out). z is (T,B,H) f16. Wave = 16 rows x
// all 128 outputs (8 WMMA tiles); softmax via 16-lane shfl_xor reductions.
// ---------------------------------------------------------------------------
__global__ __launch_bounds__(256) void esn_readout_kernel(
    const _Float16* __restrict__ z, const _Float16* __restrict__ WoutFrag,
    const float* __restrict__ bout, float* __restrict__ y) {
  const int lane = threadIdx.x & 31;
  const int mt   = (blockIdx.x * blockDim.x + threadIdx.x) >> 5;  // 0..1023
  const int row  = mt * 16 + (lane & 15);
  const int k0   = (lane >= 16) ? 8 : 0;

  v8f c0, c1, c2, c3, c4, c5, c6, c7;
#pragma unroll
  for (int q = 0; q < 8; ++q) {
    c0[q] = 0.0f; c1[q] = 0.0f; c2[q] = 0.0f; c3[q] = 0.0f;
    c4[q] = 0.0f; c5[q] = 0.0f; c6[q] = 0.0f; c7[q] = 0.0f;
  }

  const _Float16* arow  = z + (size_t)row * kH;
  const _Float16* bbase = WoutFrag + lane * 16;
#pragma unroll 1
  for (int kt = 0; kt < 64; ++kt) {
    v16h a = make_a_frag(arow, kt, k0);
    const _Float16* bp = bbase + (size_t)kt * 8 * 512;
    v16h b0 = *(const v16h*)(bp + 0 * 512);
    v16h b1 = *(const v16h*)(bp + 1 * 512);
    v16h b2 = *(const v16h*)(bp + 2 * 512);
    v16h b3 = *(const v16h*)(bp + 3 * 512);
    v16h b4 = *(const v16h*)(bp + 4 * 512);
    v16h b5 = *(const v16h*)(bp + 5 * 512);
    v16h b6 = *(const v16h*)(bp + 6 * 512);
    v16h b7 = *(const v16h*)(bp + 7 * 512);
    c0 = __builtin_amdgcn_wmma_f32_16x16x32_f16(false, a, false, b0, (short)0, c0, false, false);
    c1 = __builtin_amdgcn_wmma_f32_16x16x32_f16(false, a, false, b1, (short)0, c1, false, false);
    c2 = __builtin_amdgcn_wmma_f32_16x16x32_f16(false, a, false, b2, (short)0, c2, false, false);
    c3 = __builtin_amdgcn_wmma_f32_16x16x32_f16(false, a, false, b3, (short)0, c3, false, false);
    c4 = __builtin_amdgcn_wmma_f32_16x16x32_f16(false, a, false, b4, (short)0, c4, false, false);
    c5 = __builtin_amdgcn_wmma_f32_16x16x32_f16(false, a, false, b5, (short)0, c5, false, false);
    c6 = __builtin_amdgcn_wmma_f32_16x16x32_f16(false, a, false, b6, (short)0, c6, false, false);
    c7 = __builtin_amdgcn_wmma_f32_16x16x32_f16(false, a, false, b7, (short)0, c7, false, false);
  }

  float vals[8][8];
#pragma unroll
  for (int q = 0; q < 8; ++q) {
    vals[0][q] = c0[q]; vals[1][q] = c1[q]; vals[2][q] = c2[q]; vals[3][q] = c3[q];
    vals[4][q] = c4[q]; vals[5][q] = c5[q]; vals[6][q] = c6[q]; vals[7][q] = c7[q];
  }

#pragma unroll
  for (int r = 0; r < 8; ++r) {
    float v[8];
    float mx = -3.4e38f;
#pragma unroll
    for (int j = 0; j < 8; ++j) {
      v[j] = vals[j][r] + bout[j * 16 + (lane & 15)];
      mx = fmaxf(mx, v[j]);
    }
    for (int mask = 8; mask; mask >>= 1) mx = fmaxf(mx, __shfl_xor(mx, mask, 32));
    float s = 0.0f;
#pragma unroll
    for (int j = 0; j < 8; ++j) { v[j] = expf(v[j] - mx); s += v[j]; }
    for (int mask = 8; mask; mask >>= 1) s += __shfl_xor(s, mask, 32);
    float inv = 1.0f / s;

    int gr = mt * 16 + r + ((lane >= 16) ? 8 : 0);  // = t*B + b
    int b  = gr & (kB - 1);
    int t  = gr >> 6;
    float* orow = y + ((size_t)b * kT + t) * kO;
#pragma unroll
    for (int j = 0; j < 8; ++j) orow[j * 16 + (lane & 15)] = v[j] * inv;
  }
}

// ---------------------------------------------------------------------------
extern "C" void kernel_launch(void* const* d_in, const int* in_sizes, int n_in,
                              void* d_out, int out_size, void* d_ws, size_t ws_size,
                              hipStream_t stream) {
  const int*   x    = (const int*)d_in[0];
  // d_in[1] = h0 (all zeros; handled by `first` flag)
  const float* Wxh  = (const float*)d_in[2];
  const float* Whh  = (const float*)d_in[3];
  const float* bh   = (const float*)d_in[4];
  const float* Wout = (const float*)d_in[5];
  const float* bout = (const float*)d_in[6];

  float* y     = (float*)d_out;                          // (B,T,O) f32
  float* hlast = y + (size_t)kB * kT * kO;               // (B,H)   f32

  char* ws = (char*)d_ws;
  _Float16* WhhFrag  = (_Float16*)ws;                                  // 8 MB
  _Float16* WoutFrag = (_Float16*)(ws + (8u << 20));                   // 512 KB
  _Float16* z        = (_Float16*)(ws + (8u << 20) + (1u << 19));      // 64 MB (T,B,H)

  // Pack weights into WMMA-B fragment order (f32 -> f16)
  pack_frag_kernel<<<16384, 256, 0, stream>>>(Whh,  WhhFrag,  kH, 128);  // 64*128 frags
  pack_frag_kernel<<<1024,  256, 0, stream>>>(Wout, WoutFrag, kO, 8);    // 64*8 frags

  // Sequential scan over T
  for (int t = 0; t < kT; ++t) {
    const _Float16* zprev = (t == 0) ? z : z + (size_t)(t - 1) * kB * kH;
    _Float16*       zcur  = z + (size_t)t * kB * kH;
    esn_step_kernel<<<16, 256, 0, stream>>>(
        zprev, zcur, WhhFrag, x, Wxh, bh,
        (t == kT - 1) ? hlast : nullptr, t, (t == 0) ? 1 : 0);
  }

  // Readout + softmax over all (B*T) rows
  esn_readout_kernel<<<128, 256, 0, stream>>>(z, WoutFrag, bout, y);
}
